// DilatedAttention_24292335027124
// MI455X (gfx1250) — matrix operand order repaired and checked by reading, not compiled, not run on hardware
//
#include <hip/hip_runtime.h>
#include <hip/hip_bf16.h>

// Problem constants (from reference): B=2, S=2048, E=1024, H=16, D=64.
#define BB 2
#define SS 2048
#define EE 1024
#define HH 16
#define DD 64

typedef __attribute__((ext_vector_type(16))) __bf16 v16bf;
typedef __attribute__((ext_vector_type(8)))  float  v8f;

union FragU { v16bf v; uint4 q[2]; };

// A-matrix fragment (16x32 bf16), per ISA 7.12.2:
// lanes 0-15 : row m=lane,    K = kb+{0..7} (v0-3), kb+{16..23} (v4-7)
// lanes 16-31: row m=lane-16, K = kb+{8..15},        kb+{24..31}
__device__ __forceinline__ v16bf load_frag_a(const __bf16* base, int ld, int kbase, int lane) {
  const int m = lane & 15, hlf = lane >> 4;
  const __bf16* r = base + (size_t)m * ld + kbase + hlf * 8;
  FragU u;
  u.q[0] = *(const uint4*)(r);
  u.q[1] = *(const uint4*)(r + 16);
  return u.v;
}

// B-matrix fragment (32x16 bf16) from LDS stored [n][k] (k contiguous):
// lane n (0-15): K = kb+{0..7}, kb+{16..23}; lane 16+n: K = kb+{8..15}, kb+{24..31}, column n.
__device__ __forceinline__ v16bf load_frag_b(const __bf16* base, int n0, int ld, int kbase, int lane) {
  const int n = lane & 15, hlf = lane >> 4;
  const __bf16* r = base + (size_t)(n0 + n) * ld + kbase + hlf * 8;
  FragU u;
  u.q[0] = *(const uint4*)(r);
  u.q[1] = *(const uint4*)(r + 16);
  return u.v;
}

__device__ __forceinline__ v8f wmma_bf16(v16bf a, v16bf b, v8f c) {
  return __builtin_amdgcn_wmma_f32_16x16x32_bf16(false, a, false, b, (short)0, c, false, false);
}

struct bf16x4 { __bf16 a, b, c, d; };

__global__ void cvt_f32_bf16(const float* __restrict__ in, __bf16* __restrict__ out, int n4) {
  int i = blockIdx.x * blockDim.x + threadIdx.x;
  if (i < n4) {
    float4 f = ((const float4*)in)[i];
    bf16x4 r = { (__bf16)f.x, (__bf16)f.y, (__bf16)f.z, (__bf16)f.w };
    ((bf16x4*)out)[i] = r;
  }
}

// y[row, ncol] = sum_k A[row,k] * W[ncol,k] + bias[ncol]
// MODE 0: store bf16 to [B,H,S',D]   (Q/K, dilation-permuted)
// MODE 1: store bf16 to [B,H,D,S']   (V, permuted + transposed)
// MODE 2: store f32  to [B,S,E]      (final output)
template<int MODE>
__global__ __launch_bounds__(128) void gemm64(const __bf16* __restrict__ A,
                                              const __bf16* __restrict__ W,
                                              const float* __restrict__ bias,
                                              void* __restrict__ outp) {
  __shared__ __align__(16) __bf16 As[2][64 * 72];
  __shared__ __align__(16) __bf16 Bs[2][64 * 72];
  const int tid = threadIdx.x, lane = tid & 31, wave = tid >> 5;
  const int bm = blockIdx.x, bn = blockIdx.y;

  const v8f zero = {0.f,0.f,0.f,0.f,0.f,0.f,0.f,0.f};
  v8f acc[4] = {zero, zero, zero, zero};

  // Stage one 64x64 A tile + 64x64 W tile into LDS buffer `buf`.
  auto stage = [&](int kt, int buf) {
#pragma unroll
    for (int it = 0; it < 2; ++it) {
      int idx = tid + it * 128;          // 256 tasks: 64 rows x 4 segments
      int row = idx >> 2, seg = idx & 3;
      *(uint4*)(&As[buf][row * 72 + seg * 16]) =
          *(const uint4*)(A + (size_t)(bm * 64 + row) * EE + kt * 64 + seg * 16);
      *(uint4*)(&Bs[buf][row * 72 + seg * 16]) =
          *(const uint4*)(W + (size_t)(bn * 64 + row) * EE + kt * 64 + seg * 16);
      if (kt + 2 < EE / 64) {
        __builtin_prefetch(A + (size_t)(bm * 64 + row) * EE + (kt + 2) * 64 + seg * 16, 0, 1);
        __builtin_prefetch(W + (size_t)(bn * 64 + row) * EE + (kt + 2) * 64 + seg * 16, 0, 1);
      }
    }
  };

  stage(0, 0);
  for (int kt = 0; kt < EE / 64; ++kt) {
    __syncthreads();                       // buf kt&1 staged; buf (kt+1)&1 free
    if (kt + 1 < EE / 64) stage(kt + 1, (kt + 1) & 1);
    const __bf16* as = &As[kt & 1][0];
    const __bf16* bs = &Bs[kt & 1][0];
#pragma unroll
    for (int kb = 0; kb < 64; kb += 32) {
      v16bf af = load_frag_a(as + wave * 16 * 72, 72, kb, lane);
#pragma unroll
      for (int nt = 0; nt < 4; ++nt) {
        v16bf bfrag = load_frag_b(bs, nt * 16, 72, kb, lane);
        acc[nt] = wmma_bf16(af, bfrag, acc[nt]);
      }
    }
  }

  const int hlf = lane >> 4, nlo = lane & 15;
#pragma unroll
  for (int nt = 0; nt < 4; ++nt) {
    const int ncol = bn * 64 + nt * 16 + nlo;
    const float bv = bias[ncol];
    const int h = ncol >> 6, c = ncol & 63;
    const int ldil = h & 3;                       // dilation = 1 << (h%4)
#pragma unroll
    for (int p = 0; p < 8; ++p) {
      const int row = bm * 64 + wave * 16 + p + 8 * hlf;  // = b*S + s
      const float val = acc[nt][p] + bv;
      if (MODE == 2) {
        ((float*)outp)[(size_t)row * EE + ncol] = val;
      } else {
        const int b = row >> 11, s = row & (SS - 1);
        const int phase = s & ((1 << ldil) - 1);
        const int s2 = phase * (SS >> ldil) + (s >> ldil);   // permuted index
        if (MODE == 0)
          ((__bf16*)outp)[(((size_t)b * HH + h) * SS + s2) * DD + c] = (__bf16)val;
        else
          ((__bf16*)outp)[(((size_t)b * HH + h) * DD + c) * SS + s2] = (__bf16)val;
      }
    }
  }
}

// Flash attention over the permuted sequence: per (b,h), each 64-row query tile
// attends densely to its own phase group [g0, g0+Lg). No masking needed.
__global__ __launch_bounds__(128) void dil_attn(const __bf16* __restrict__ q,
                                                const __bf16* __restrict__ k,
                                                const __bf16* __restrict__ vT,
                                                __bf16* __restrict__ attn) {
  __shared__ __align__(16) __bf16 Ks[2][64 * 72];  // [key][ch]  -> B-operand for QK^T
  __shared__ __align__(16) __bf16 Vs[2][64 * 72];  // [ch][key]  -> B-operand for PV
  __shared__ __align__(16) __bf16 Ps[4][16 * 72];  // per-wave P tile [m][t]

  const int tid = threadIdx.x, lane = tid & 31, wave = tid >> 5;
  const int hlf = lane >> 4, nlo = lane & 15;
  const int bh = blockIdx.y;
  const int b = bh >> 4, h = bh & 15;
  const int ldil = h & 3;
  const int Lg = SS >> ldil;              // phase-group length (>= 256, mult of 64)
  const int s0 = blockIdx.x * 64;         // query tile base (permuted space)
  const int g0 = (s0 / Lg) * Lg;          // group key start

  // Q fragments held in registers for the whole KV loop.
  const __bf16* qrow = q + ((size_t)bh * SS + s0 + wave * 16) * DD;
  const v16bf qf0 = load_frag_a(qrow, DD, 0, lane);
  const v16bf qf1 = load_frag_a(qrow, DD, 32, lane);

  const v8f zero = {0.f,0.f,0.f,0.f,0.f,0.f,0.f,0.f};
  v8f o[4] = {zero, zero, zero, zero};
  float mr[8], lr[8];
#pragma unroll
  for (int p = 0; p < 8; ++p) { mr[p] = -1e30f; lr[p] = 0.f; }

  auto stageKV = [&](int kt0, int buf) {
#pragma unroll
    for (int it = 0; it < 2; ++it) {
      int idx = tid + it * 128;
      int row = idx >> 2, seg = idx & 3;
      *(uint4*)(&Ks[buf][row * 72 + seg * 16]) =
          *(const uint4*)(k + ((size_t)bh * SS + kt0 + row) * DD + seg * 16);
      *(uint4*)(&Vs[buf][row * 72 + seg * 16]) =
          *(const uint4*)(vT + ((size_t)bh * DD + row) * SS + kt0 + seg * 16);
    }
  };

  const int nkt = Lg >> 6;
  stageKV(g0, 0);
  for (int ktile = 0; ktile < nkt; ++ktile) {
    __syncthreads();                       // buf ktile&1 staged; other buf free
    if (ktile + 1 < nkt) stageKV(g0 + (ktile + 1) * 64, (ktile + 1) & 1);
    const __bf16* ks = &Ks[ktile & 1][0];
    const __bf16* vs = &Vs[ktile & 1][0];

    // scores = Q K^T  (16 query rows x 64 keys per wave)
    v8f sc[4] = {zero, zero, zero, zero};
#pragma unroll
    for (int nt = 0; nt < 4; ++nt) {
      sc[nt] = wmma_bf16(qf0, load_frag_b(ks, nt * 16, 72, 0, lane), sc[nt]);
      sc[nt] = wmma_bf16(qf1, load_frag_b(ks, nt * 16, 72, 32, lane), sc[nt]);
    }

    // online softmax: rows live across 16-lane halves in C-layout
    float corr[8], rs[8];
#pragma unroll
    for (int p = 0; p < 8; ++p) {
      float x = fmaxf(fmaxf(sc[0][p], sc[1][p]), fmaxf(sc[2][p], sc[3][p])) * 0.125f;
      x = fmaxf(x, __shfl_xor(x, 1, 32));
      x = fmaxf(x, __shfl_xor(x, 2, 32));
      x = fmaxf(x, __shfl_xor(x, 4, 32));
      x = fmaxf(x, __shfl_xor(x, 8, 32));
      const float mn = fmaxf(mr[p], x);
      corr[p] = __expf(mr[p] - mn);
      mr[p] = mn;
      rs[p] = 0.f;
    }

    __bf16* pw = &Ps[wave][0];
#pragma unroll
    for (int nt = 0; nt < 4; ++nt) {
#pragma unroll
      for (int p = 0; p < 8; ++p) {
        const float e = __expf(sc[nt][p] * 0.125f - mr[p]);
        rs[p] += e;
        pw[(p + 8 * hlf) * 72 + nt * 16 + nlo] = (__bf16)e;  // P[m][t]
      }
    }
#pragma unroll
    for (int p = 0; p < 8; ++p) {
      float r = rs[p];
      r += __shfl_xor(r, 1, 32);
      r += __shfl_xor(r, 2, 32);
      r += __shfl_xor(r, 4, 32);
      r += __shfl_xor(r, 8, 32);
      lr[p] = lr[p] * corr[p] + r;
#pragma unroll
      for (int nt = 0; nt < 4; ++nt) o[nt][p] *= corr[p];
    }

    // O += P V  (P re-read from per-wave LDS as A fragments; in-wave DS is in-order)
#pragma unroll
    for (int kb = 0; kb < 64; kb += 32) {
      v16bf pf = load_frag_a(pw, 72, kb, lane);
#pragma unroll
      for (int nt = 0; nt < 4; ++nt)
        o[nt] = wmma_bf16(pf, load_frag_b(vs, nt * 16, 72, kb, lane), o[nt]);
    }
  }

  // normalize + un-permute store to attn[b, s, h*64 + c]
#pragma unroll
  for (int nt = 0; nt < 4; ++nt) {
    const int c = nt * 16 + nlo;
#pragma unroll
    for (int p = 0; p < 8; ++p) {
      const int sp = s0 + wave * 16 + p + 8 * hlf;       // permuted s'
      const int phase = sp >> (11 - ldil);
      const int t = sp & (Lg - 1);
      const int s = phase + (t << ldil);                 // original position
      attn[((size_t)b * SS + s) * EE + h * 64 + c] = (__bf16)(o[nt][p] / lr[p]);
    }
  }
}

extern "C" void kernel_launch(void* const* d_in, const int* in_sizes, int n_in,
                              void* d_out, int out_size, void* d_ws, size_t ws_size,
                              hipStream_t stream) {
  (void)in_sizes; (void)n_in; (void)out_size; (void)ws_size;
  const float* x  = (const float*)d_in[0];
  const float* Wq = (const float*)d_in[1];
  const float* bq = (const float*)d_in[2];
  const float* Wk = (const float*)d_in[3];
  const float* bk = (const float*)d_in[4];
  const float* Wv = (const float*)d_in[5];
  const float* bv = (const float*)d_in[6];
  const float* Wo = (const float*)d_in[7];
  const float* bo = (const float*)d_in[8];

  const size_t nx = (size_t)BB * SS * EE;  // 4M elements
  const size_t nw = (size_t)EE * EE;       // 1M elements

  // Workspace layout (40 MB): attn reuses the x_bf region once x is dead.
  char* ws = (char*)d_ws;
  __bf16* x_bf  = (__bf16*)(ws);                       //  8 MB
  __bf16* wq_bf = (__bf16*)(ws + (8u << 20));          //  2 MB
  __bf16* wk_bf = wq_bf + nw;                          //  2 MB
  __bf16* wv_bf = wk_bf + nw;                          //  2 MB
  __bf16* wo_bf = wv_bf + nw;                          //  2 MB
  __bf16* qb    = wo_bf + nw;                          //  8 MB  [B,H,S',D]
  __bf16* kbuf  = qb + nx;                             //  8 MB  [B,H,S',D]
  __bf16* vTb   = kbuf + nx;                           //  8 MB  [B,H,D,S']
  __bf16* attn  = x_bf;                                //  aliases x_bf (dead after V GEMM)

  // fp32 -> bf16 conversions
  cvt_f32_bf16<<<dim3((unsigned)(nx / 4 / 256)), dim3(256), 0, stream>>>(x,  x_bf,  (int)(nx / 4));
  cvt_f32_bf16<<<dim3((unsigned)(nw / 4 / 256)), dim3(256), 0, stream>>>(Wq, wq_bf, (int)(nw / 4));
  cvt_f32_bf16<<<dim3((unsigned)(nw / 4 / 256)), dim3(256), 0, stream>>>(Wk, wk_bf, (int)(nw / 4));
  cvt_f32_bf16<<<dim3((unsigned)(nw / 4 / 256)), dim3(256), 0, stream>>>(Wv, wv_bf, (int)(nw / 4));
  cvt_f32_bf16<<<dim3((unsigned)(nw / 4 / 256)), dim3(256), 0, stream>>>(Wo, wo_bf, (int)(nw / 4));

  const dim3 gg(64, 16), tb(128);
  gemm64<0><<<gg, tb, 0, stream>>>(x_bf, wq_bf, bq, qb);
  gemm64<0><<<gg, tb, 0, stream>>>(x_bf, wk_bf, bk, kbuf);
  gemm64<1><<<gg, tb, 0, stream>>>(x_bf, wv_bf, bv, vTb);

  dil_attn<<<dim3(32, BB * HH), tb, 0, stream>>>(qb, kbuf, vTb, attn);

  gemm64<2><<<gg, tb, 0, stream>>>(attn, wo_bf, bo, d_out);
}
